// NeuralLongTermMemory_47038481825923
// MI455X (gfx1250) — matrix operand order repaired and checked by reading, not compiled, not run on hardware
//
#include <hip/hip_runtime.h>
#include <hip/hip_bf16.h>

// ---------------- types ----------------
typedef __bf16 bf16_t;
typedef bf16_t bf16x8  __attribute__((ext_vector_type(8)));
typedef bf16_t bf16x16 __attribute__((ext_vector_type(16)));
typedef float  f32x8   __attribute__((ext_vector_type(8)));

#define DEV static __device__ __forceinline__

static constexpr int Bc = 8, Sc = 2048, Dc = 1024, Hc = 16;
static constexpr int HDc = 64, HIDc = 256;
static constexpr int Nrows = Bc * Sc;   // 16384
static constexpr int BHc = Bc * Hc;     // 128

DEV f32x8 vzero8() { f32x8 z; for (int i = 0; i < 8; i++) z[i] = 0.f; return z; }

DEV f32x8 wmma_bf16(bf16x16 a, bf16x16 b, f32x8 c) {
  return __builtin_amdgcn_wmma_f32_16x16x32_bf16(
      false, a, false, b, (short)0, c, false, false);
}

// Flat (generic) pointer to raw LDS offset via explicit addrspace(3) cast.
DEV unsigned lds_addr_of(const void* p) {
  return (unsigned)(size_t)(__attribute__((address_space(3))) const char*)p;
}

// Async global -> LDS copy, 16 bytes per lane (CDNA5 ASYNCcnt path).
DEV void async_b128(unsigned ldsdst, const void* g) {
  asm volatile("global_load_async_to_lds_b128 %0, %1, off"
               :: "v"(ldsdst), "v"(g) : "memory");
}
DEV void wait_async0() {
  asm volatile("s_wait_asynccnt 0" ::: "memory");
}

// A-fragment 16x32 from row-major LDS tile (row stride ldk elements).
// lanes 0-15: m=m0+lane, K = k0..k0+15
// lanes 16-31: m=m0+lane-16, K = k0+16..k0+31
DEV bf16x16 load_frag_a(const bf16_t* lds, int m0, int k0, int ldk, int lane) {
  int half = lane >> 4;
  int m = m0 + (lane & 15);
  const bf16_t* p = lds + m * ldk + k0 + half * 16;
  bf16x8 lo = *(const bf16x8*)p;
  bf16x8 hi = *(const bf16x8*)(p + 8);
  bf16x16 r;
  for (int i = 0; i < 8; i++) { r[i] = lo[i]; r[8 + i] = hi[i]; }
  return r;
}

// B-fragment 32x16 from LDS holding B^T row-major [N][K] (row stride ldk).
// lanes 0-15: n=n0+lane, K=k0..k0+15; lanes 16-31: n=n0+lane-16, K=k0+16..k0+31
DEV bf16x16 load_frag_b(const bf16_t* ldsBT, int n0, int k0, int ldk, int lane) {
  int half = lane >> 4;
  int n = n0 + (lane & 15);
  return *(const bf16x16*)(ldsBT + n * ldk + k0 + half * 16);
}

// B-fragment when only B row-major [K][N] is in LDS: BT[n][k] = W[k][n] (gather).
DEV bf16x16 load_frag_b_gather(const bf16_t* W, int n0, int k0, int ldn, int lane) {
  int half = lane >> 4;
  int n = n0 + (lane & 15);
  int kb = k0 + half * 16;
  bf16x16 r;
  for (int i = 0; i < 16; i++) r[i] = W[(kb + i) * ldn + n];
  return r;
}

DEV float gelu_f(float x) { return 0.5f * x * (1.f + erff(x * 0.70710678118654752f)); }
DEV float gelu_g(float x) {
  float cdf = 0.5f * (1.f + erff(x * 0.70710678118654752f));
  float pdf = __expf(-0.5f * x * x) * 0.3989422804014327f;
  return cdf + x * pdf;
}

// ---------------- row RMSNorm (learned weight, eps 1e-6) -> bf16 ----------------
__global__ __launch_bounds__(256) void k_rmsnorm_row(
    const float* __restrict__ x, const float* __restrict__ w, bf16_t* __restrict__ out) {
  __shared__ float red[256];
  int row = blockIdx.x, t = threadIdx.x;
  const float* xr = x + (size_t)row * Dc;
  float v[4], ss = 0.f;
  for (int i = 0; i < 4; i++) { v[i] = xr[t + i * 256]; ss += v[i] * v[i]; }
  red[t] = ss; __syncthreads();
  for (int off = 128; off; off >>= 1) { if (t < off) red[t] += red[t + off]; __syncthreads(); }
  float inv = rsqrtf(red[0] * (1.f / Dc) + 1e-6f);
  bf16_t* orow = out + (size_t)row * Dc;
  for (int i = 0; i < 4; i++) { int c = t + i * 256; orow[c] = (bf16_t)(v[i] * inv * w[c]); }
}

// ---------------- f32 W[K][N] -> bf16 W^T[N][K] ----------------
__global__ __launch_bounds__(256) void k_convT(
    const float* __restrict__ W, bf16_t* __restrict__ WT, int K, int Nn) {
  size_t i = (size_t)blockIdx.x * 256 + threadIdx.x;
  if (i < (size_t)K * Nn) {
    int k = (int)(i / Nn), n = (int)(i % Nn);
    WT[(size_t)n * K + k] = (bf16_t)W[i];
  }
}

// ------- big bf16 WMMA GEMM, double-buffered async global->LDS: C = A * B, BT=[N][K] -------
__global__ __launch_bounds__(256) void k_gemm_bf16(
    const bf16_t* __restrict__ A, const bf16_t* __restrict__ BT,
    float* __restrict__ C, int M, int Nn, int K) {
  __shared__ alignas(32) bf16_t sA[2][128][32];
  __shared__ alignas(32) bf16_t sB[2][128][32];
  int bm = blockIdx.x * 128, bn = blockIdx.y * 128;
  int tid = threadIdx.x, lane = tid & 31, wv = tid >> 5;
  int wm = (wv >> 2) * 64, wn = (wv & 3) * 32;
  f32x8 acc[4][2];
  for (int i = 0; i < 4; i++) for (int j = 0; j < 2; j++) acc[i][j] = vzero8();

  int r = tid >> 1, ch = (tid & 1) * 16;
  const bf16_t* gA = A + (size_t)(bm + r) * K + ch;
  const bf16_t* gB = BT + (size_t)(bn + r) * K + ch;

  auto issue = [&](int buf, int k0) {
    unsigned la = lds_addr_of(&sA[buf][r][ch]);
    unsigned lb = lds_addr_of(&sB[buf][r][ch]);
    async_b128(la,      gA + k0);
    async_b128(la + 16, gA + k0 + 8);
    async_b128(lb,      gB + k0);
    async_b128(lb + 16, gB + k0 + 8);
  };

  issue(0, 0);
  wait_async0();
  __syncthreads();

  int buf = 0;
  for (int k0 = 0; k0 < K; k0 += 32) {
    if (k0 + 32 < K) issue(buf ^ 1, k0 + 32);
    bf16x16 af[4], bfv[2];
    for (int i = 0; i < 4; i++) af[i]  = load_frag_a(&sA[buf][0][0], wm + i * 16, 0, 32, lane);
    for (int j = 0; j < 2; j++) bfv[j] = load_frag_b(&sB[buf][0][0], wn + j * 16, 0, 32, lane);
    for (int i = 0; i < 4; i++)
      for (int j = 0; j < 2; j++)
        acc[i][j] = wmma_bf16(af[i], bfv[j], acc[i][j]);
    wait_async0();
    __syncthreads();
    buf ^= 1;
  }

  int half = lane >> 4, col = lane & 15;
  for (int i = 0; i < 4; i++)
    for (int j = 0; j < 2; j++)
      for (int e = 0; e < 8; e++) {
        int row = bm + wm + i * 16 + half * 8 + e;
        int cc = bn + wn + j * 16 + col;
        C[(size_t)row * Nn + cc] = acc[i][j][e];
      }
}

// ---------------- split heads (+optional head RMSNorm, eps 1e-6) -> bf16 (BH,S,HD) ----------------
__global__ __launch_bounds__(64) void k_split(
    const float* __restrict__ C, const float* __restrict__ normw, bf16_t* __restrict__ out) {
  __shared__ float red[64];
  int bs = blockIdx.x, h = blockIdx.y, t = threadIdx.x;
  int b = bs >> 11, s = bs & (Sc - 1);
  float v = C[(size_t)bs * Dc + h * HDc + t];
  if (normw) {
    red[t] = v * v; __syncthreads();
    for (int off = 32; off; off >>= 1) { if (t < off) red[t] += red[t + off]; __syncthreads(); }
    v = v * rsqrtf(red[0] * (1.f / HDc) + 1e-6f) * normw[t];
  }
  out[(((size_t)(b * Hc + h)) * Sc + s) * HDc + t] = (bf16_t)v;
}

// ---------------- gates: alpha/theta/eta (BH,S) ----------------
__global__ __launch_bounds__(64) void k_gates(
    const bf16_t* __restrict__ xsn,
    const float* __restrict__ aw, const float* __restrict__ ab,
    const float* __restrict__ tw, const float* __restrict__ tb,
    const float* __restrict__ ew, const float* __restrict__ eb,
    float* __restrict__ alpha, float* __restrict__ theta, float* __restrict__ eta) {
  __shared__ float row[Dc];
  int bs = blockIdx.x, t = threadIdx.x;
  const bf16_t* xr = xsn + (size_t)bs * Dc;
  for (int i = t; i < Dc; i += 64) row[i] = (float)xr[i];
  __syncthreads();
  if (t < 48) {
    int g = t >> 4, h = t & 15;
    const float* w = (g == 0) ? aw : (g == 1) ? tw : ew;
    const float* bv = (g == 0) ? ab : (g == 1) ? tb : eb;
    float s = 0.f;
    for (int k = 0; k < Dc; k++) s += row[k] * w[k * Hc + h];
    s += bv[h];
    float sig = 1.f / (1.f + __expf(-s));
    int b = bs >> 11, ss = bs & (Sc - 1);
    size_t idx = ((size_t)(b * Hc + h)) * Sc + ss;
    if (g == 0) alpha[idx] = sig;
    else if (g == 1) theta[idx] = sig * 0.01f;  // MAX_LR
    else eta[idx] = sig;
  }
}

// ---------------- per-bh means over S ----------------
__global__ __launch_bounds__(256) void k_mean(
    const float* __restrict__ alpha, const float* __restrict__ eta,
    float* __restrict__ amean, float* __restrict__ emean) {
  __shared__ float sa[256], se[256];
  int bh = blockIdx.x, t = threadIdx.x;
  float a = 0.f, e = 0.f;
  for (int s = t; s < Sc; s += 256) { a += alpha[(size_t)bh * Sc + s]; e += eta[(size_t)bh * Sc + s]; }
  sa[t] = a; se[t] = e; __syncthreads();
  for (int off = 128; off; off >>= 1) {
    if (t < off) { sa[t] += sa[t + off]; se[t] += se[t + off]; }
    __syncthreads();
  }
  if (t == 0) { amean[bh] = sa[0] * (1.f / Sc); emean[bh] = se[0] * (1.f / Sc); }
}

// ---------------- fused per-head memory grad + clip + weight update ----------------
__global__ __launch_bounds__(256) void k_memgrad(
    const bf16_t* __restrict__ keys, const bf16_t* __restrict__ vals,
    const float* __restrict__ theta, const float* __restrict__ W0g,
    const float* __restrict__ W1g, const float* __restrict__ amean,
    bf16_t* __restrict__ nW0T, bf16_t* __restrict__ nW1T) {
  __shared__ alignas(32) bf16_t sW0T[HIDc][HDc];   // BT for layer0 (W0^T)
  __shared__ alignas(32) bf16_t sW1 [HIDc][HDc];   // W1 row-major == BT for d_h1@W1^T
  __shared__ alignas(32) bf16_t sK  [32][HDc];
  __shared__ alignas(32) bf16_t sKT [HDc][32];
  __shared__ alignas(32) bf16_t sV  [32][HDc];
  __shared__ float sTh[32];
  __shared__ alignas(32) bf16_t sPre0[32][HIDc];
  __shared__ alignas(32) bf16_t sH1 [32][HIDc];
  __shared__ alignas(32) bf16_t sH1T[HIDc][32];
  __shared__ float sPre1[32][HDc];
  __shared__ alignas(32) bf16_t sDh1 [32][HDc];
  __shared__ alignas(32) bf16_t sDh1T[HDc][32];
  __shared__ alignas(32) bf16_t sDh0T[HIDc][32];
  __shared__ float sRedF[256];

  int tid = threadIdx.x, lane = tid & 31, w = tid >> 5;
  int half = lane >> 4, lcol = lane & 15;
  int bh = blockIdx.x;

  for (int idx = tid; idx < HDc * HIDc; idx += 256) {
    int k = idx >> 8, n = idx & (HIDc - 1);
    sW0T[n][k] = (bf16_t)W0g[idx];                  // W0[k][n] -> W0T[n][k]
    sW1[idx >> 6][idx & 63] = (bf16_t)W1g[idx];     // W1 row-major [256][64]
  }
  __syncthreads();

  f32x8 accG0[8], accG1[8];
  for (int i = 0; i < 8; i++) { accG0[i] = vzero8(); accG1[i] = vzero8(); }

  const size_t rowbase = (size_t)bh * Sc;

  for (int s0 = 0; s0 < Sc; s0 += 32) {
    for (int idx = tid; idx < 32 * HDc; idx += 256) {
      int r = idx >> 6, c = idx & 63;
      size_t g = (rowbase + s0 + r) * HDc + c;
      bf16_t kv = keys[g];
      sK[r][c] = kv; sKT[c][r] = kv; sV[r][c] = vals[g];
    }
    if (tid < 32) sTh[tid] = theta[rowbase + s0 + tid];
    __syncthreads();

    // pre0 = K @ W0  (M=32,N=256,K=64); h1 = gelu(pre0)
    {
      int mt = w & 1, ng = w >> 1;
      f32x8 p[4]; for (int j = 0; j < 4; j++) p[j] = vzero8();
      for (int kk = 0; kk < HDc; kk += 32) {
        bf16x16 a = load_frag_a(&sK[0][0], mt * 16, kk, HDc, lane);
        for (int j = 0; j < 4; j++) {
          bf16x16 b = load_frag_b(&sW0T[0][0], (ng * 4 + j) * 16, kk, HDc, lane);
          p[j] = wmma_bf16(a, b, p[j]);
        }
      }
      for (int j = 0; j < 4; j++)
        for (int r = 0; r < 8; r++) {
          int row = mt * 16 + half * 8 + r;
          int col = (ng * 4 + j) * 16 + lcol;
          float pv = p[j][r];
          sPre0[row][col] = (bf16_t)pv;
          float gv = gelu_f(pv);
          sH1[row][col] = (bf16_t)gv;
          sH1T[col][row] = (bf16_t)gv;
        }
    }
    __syncthreads();

    // pre1 = h1 @ W1 (M=32,N=64,K=256)
    {
      int mt = w >> 2, nt = w & 3;
      f32x8 p = vzero8();
      for (int kk = 0; kk < HIDc; kk += 32) {
        bf16x16 a = load_frag_a(&sH1[0][0], mt * 16, kk, HIDc, lane);
        bf16x16 b = load_frag_b_gather(&sW1[0][0], nt * 16, kk, HDc, lane);
        p = wmma_bf16(a, b, p);
      }
      for (int r = 0; r < 8; r++)
        sPre1[mt * 16 + half * 8 + r][nt * 16 + lcol] = p[r];
    }
    __syncthreads();

    // row-wise: pf-RMSNorm, pred, d_pred, d_h1 (norm backward)
    {
      int r = tid >> 3, j = tid & 7;
      float hv[8], hn[8], dp[8], ss = 0.f;
      for (int i = 0; i < 8; i++) { float p = sPre1[r][j * 8 + i]; hv[i] = p; ss += p * p; }
      sRedF[tid] = ss; __syncthreads();
      if (j < 4) sRedF[tid] += sRedF[tid + 4]; __syncthreads();
      if (j < 2) sRedF[tid] += sRedF[tid + 2]; __syncthreads();
      if (j < 1) sRedF[tid] += sRedF[tid + 1]; __syncthreads();
      float rms = sqrtf(sRedF[r * 8] * (1.f / HDc) + 1e-8f);
      float th = sTh[r];
      __syncthreads();
      float dy = 0.f;
      for (int i = 0; i < 8; i++) {
        hn[i] = hv[i] / rms;
        float pred = hn[i] + (float)sK[r][j * 8 + i];
        float d = 2.f * th * (pred - (float)sV[r][j * 8 + i]);
        dp[i] = d; dy += d * hn[i];
      }
      sRedF[tid] = dy; __syncthreads();
      if (j < 4) sRedF[tid] += sRedF[tid + 4]; __syncthreads();
      if (j < 2) sRedF[tid] += sRedF[tid + 2]; __syncthreads();
      if (j < 1) sRedF[tid] += sRedF[tid + 1]; __syncthreads();
      float dys = sRedF[r * 8];
      for (int i = 0; i < 8; i++) {
        float dh = (dp[i] - hn[i] * dys * (1.f / HDc)) / rms;
        int c = j * 8 + i;
        sDh1[r][c] = (bf16_t)dh;
        sDh1T[c][r] = (bf16_t)dh;
      }
    }
    __syncthreads();

    // G1 += h1^T @ d_h1  (M=256,N=64,K=32)
    for (int it = 0; it < 2; it++) {
      bf16x16 a = load_frag_a(&sH1T[0][0], (w * 2 + it) * 16, 0, 32, lane);
      for (int jn = 0; jn < 4; jn++) {
        bf16x16 b = load_frag_b(&sDh1T[0][0], jn * 16, 0, 32, lane);
        accG1[it * 4 + jn] = wmma_bf16(a, b, accG1[it * 4 + jn]);
      }
    }

    // d_h0 = (d_h1 @ W1^T) * gelu'(pre0)  (M=32,N=256,K=64)
    {
      int mt = w & 1, ng = w >> 1;
      f32x8 q[4]; for (int j = 0; j < 4; j++) q[j] = vzero8();
      for (int kk = 0; kk < HDc; kk += 32) {
        bf16x16 a = load_frag_a(&sDh1[0][0], mt * 16, kk, HDc, lane);
        for (int j = 0; j < 4; j++) {
          bf16x16 b = load_frag_b(&sW1[0][0], (ng * 4 + j) * 16, kk, HDc, lane);
          q[j] = wmma_bf16(a, b, q[j]);
        }
      }
      for (int j = 0; j < 4; j++)
        for (int r = 0; r < 8; r++) {
          int row = mt * 16 + half * 8 + r;
          int col = (ng * 4 + j) * 16 + lcol;
          float dh0 = q[j][r] * gelu_g((float)sPre0[row][col]);
          sDh0T[col][row] = (bf16_t)dh0;
        }
    }
    __syncthreads();

    // G0 += K^T @ d_h0  (M=64,N=256,K=32)
    {
      bf16x16 a = load_frag_a(&sKT[0][0], (w & 3) * 16, 0, 32, lane);
      for (int jt = 0; jt < 8; jt++) {
        bf16x16 b = load_frag_b(&sDh0T[0][0], ((w >> 2) * 8 + jt) * 16, 0, 32, lane);
        accG0[jt] = wmma_bf16(a, b, accG0[jt]);
      }
    }
    __syncthreads();
  }

  // grad clip (global norm over both grads) + update + transposed bf16 writeout
  float local = 0.f;
  for (int t = 0; t < 8; t++)
    for (int r = 0; r < 8; r++) {
      float a = accG0[t][r], b = accG1[t][r];
      local += a * a + b * b;
    }
  sRedF[tid] = local; __syncthreads();
  for (int off = 128; off; off >>= 1) { if (tid < off) sRedF[tid] += sRedF[tid + off]; __syncthreads(); }
  float coef = fminf(10.f / (sqrtf(sRedF[0]) + 1e-6f), 1.f);
  float one_a = 1.f - amean[bh];

  bf16_t* out0 = nW0T + (size_t)bh * HDc * HIDc;
  for (int jt = 0; jt < 8; jt++) {
    int n = ((w >> 2) * 8 + jt) * 16 + lcol;
    for (int r = 0; r < 8; r++) {
      int m = (w & 3) * 16 + half * 8 + r;
      float g = accG0[jt][r] * coef;
      float w0 = W0g[m * HIDc + n];
      float nw = one_a * w0 - g;
      if (!(fabsf(nw) < 3.0e38f)) nw = w0;
      out0[(size_t)n * HDc + m] = (bf16_t)nw;   // [256][64] = BT for retrieve layer0
    }
  }
  bf16_t* out1 = nW1T + (size_t)bh * HDc * HIDc;
  for (int it = 0; it < 2; it++) {
    for (int jn = 0; jn < 4; jn++) {
      int n = jn * 16 + lcol;
      for (int r = 0; r < 8; r++) {
        int m = (w * 2 + it) * 16 + half * 8 + r;
        float g = accG1[it * 4 + jn][r] * coef;
        float w1 = W1g[m * HDc + n];
        float nw = one_a * w1 - g;
        if (!(fabsf(nw) < 3.0e38f)) nw = w1;
        out1[(size_t)n * HIDc + m] = (bf16_t)nw; // [64][256] = BT for retrieve layer1
      }
    }
  }
}

// ---------------- retrieve: out = pf_rmsnorm(nan_to_num(MLP(q))) + q, merged heads ----------------
__global__ __launch_bounds__(256) void k_retrieve(
    const bf16_t* __restrict__ qn, const bf16_t* __restrict__ nW0T,
    const bf16_t* __restrict__ nW1T, float* __restrict__ out) {
  __shared__ alignas(32) bf16_t sW0T[HIDc][HDc];
  __shared__ alignas(32) bf16_t sW1T[HDc][HIDc];
  __shared__ alignas(32) bf16_t sQ[32][HDc];
  __shared__ alignas(32) bf16_t sH1[32][HIDc];
  __shared__ float sPre1[32][HDc];
  __shared__ float sRedF[256];

  int tid = threadIdx.x, lane = tid & 31, w = tid >> 5;
  int half = lane >> 4, lcol = lane & 15;
  int bh = blockIdx.y, s0 = blockIdx.x * 32;
  const bf16_t* w0 = nW0T + (size_t)bh * HDc * HIDc;
  const bf16_t* w1 = nW1T + (size_t)bh * HDc * HIDc;
  for (int idx = tid; idx < HDc * HIDc; idx += 256) {
    (&sW0T[0][0])[idx] = w0[idx];
    (&sW1T[0][0])[idx] = w1[idx];
  }
  for (int idx = tid; idx < 32 * HDc; idx += 256) {
    int r = idx >> 6, c = idx & 63;
    sQ[r][c] = qn[((size_t)bh * Sc + s0 + r) * HDc + c];
  }
  __syncthreads();

  // layer0: h1 = gelu(q @ W0')
  {
    int mt = w & 1, ng = w >> 1;
    f32x8 p[4]; for (int j = 0; j < 4; j++) p[j] = vzero8();
    for (int kk = 0; kk < HDc; kk += 32) {
      bf16x16 a = load_frag_a(&sQ[0][0], mt * 16, kk, HDc, lane);
      for (int j = 0; j < 4; j++) {
        bf16x16 b = load_frag_b(&sW0T[0][0], (ng * 4 + j) * 16, kk, HDc, lane);
        p[j] = wmma_bf16(a, b, p[j]);
      }
    }
    for (int j = 0; j < 4; j++)
      for (int r = 0; r < 8; r++)
        sH1[mt * 16 + half * 8 + r][(ng * 4 + j) * 16 + lcol] = (bf16_t)gelu_f(p[j][r]);
  }
  __syncthreads();

  // layer1: pre1 = h1 @ W1'
  {
    int mt = w >> 2, nt = w & 3;
    f32x8 p = vzero8();
    for (int kk = 0; kk < HIDc; kk += 32) {
      bf16x16 a = load_frag_a(&sH1[0][0], mt * 16, kk, HIDc, lane);
      bf16x16 b = load_frag_b(&sW1T[0][0], nt * 16, kk, HIDc, lane);
      p = wmma_bf16(a, b, p);
    }
    for (int r = 0; r < 8; r++)
      sPre1[mt * 16 + half * 8 + r][nt * 16 + lcol] = p[r];
  }
  __syncthreads();

  // nan_to_num + pf-RMSNorm + residual + merged-head store
  {
    int r = tid >> 3, j = tid & 7;
    float hv[8], ss = 0.f;
    for (int i = 0; i < 8; i++) {
      float h = sPre1[r][j * 8 + i];
      if (!(fabsf(h) < 3.0e38f)) h = 0.f;
      hv[i] = h; ss += h * h;
    }
    sRedF[tid] = ss; __syncthreads();
    if (j < 4) sRedF[tid] += sRedF[tid + 4]; __syncthreads();
    if (j < 2) sRedF[tid] += sRedF[tid + 2]; __syncthreads();
    if (j < 1) sRedF[tid] += sRedF[tid + 1]; __syncthreads();
    float rms = sqrtf(sRedF[r * 8] * (1.f / HDc) + 1e-8f);
    int b = bh >> 4, head = bh & 15, s = s0 + r;
    float* orow = out + ((size_t)b * Sc + s) * Dc + head * HDc;
    for (int i = 0; i < 8; i++)
      orow[j * 8 + i] = hv[i] / rms + (float)sQ[r][j * 8 + i];
  }
}

// ---------------- launcher ----------------
extern "C" void kernel_launch(void* const* d_in, const int* in_sizes, int n_in,
                              void* d_out, int out_size, void* d_ws, size_t ws_size,
                              hipStream_t stream) {
  (void)in_sizes; (void)n_in; (void)out_size; (void)ws_size;
  const float* x   = (const float*)d_in[0];
  const float* W_K = (const float*)d_in[1];
  const float* W_V = (const float*)d_in[2];
  const float* W_Q = (const float*)d_in[3];
  const float* mW0 = (const float*)d_in[4];
  const float* mW1 = (const float*)d_in[5];
  const float* knw = (const float*)d_in[6];
  const float* qnw = (const float*)d_in[7];
  const float* snw = (const float*)d_in[8];
  const float* rnw = (const float*)d_in[9];
  const float* aw  = (const float*)d_in[10];
  const float* ab  = (const float*)d_in[11];
  const float* tw  = (const float*)d_in[12];
  const float* tb  = (const float*)d_in[13];
  const float* ew  = (const float*)d_in[14];
  const float* eb  = (const float*)d_in[15];
  float* out = (float*)d_out;

  char* p = (char*)d_ws;
  auto alloc = [&](size_t bytes) -> char* {
    char* r = p; p += (bytes + 255) & ~(size_t)255; return r;
  };
  bf16_t* xsn   = (bf16_t*)alloc((size_t)Nrows * Dc * 2);
  bf16_t* wt    = (bf16_t*)alloc((size_t)Dc * Dc * 2);
  float*  Cf    = (float*) alloc((size_t)Nrows * Dc * 4);
  bf16_t* keys  = (bf16_t*)alloc((size_t)BHc * Sc * HDc * 2);
  bf16_t* vals  = (bf16_t*)alloc((size_t)BHc * Sc * HDc * 2);
  bf16_t* qn    = (bf16_t*)alloc((size_t)BHc * Sc * HDc * 2);
  float*  alp   = (float*) alloc((size_t)BHc * Sc * 4);
  float*  tht   = (float*) alloc((size_t)BHc * Sc * 4);
  float*  eta   = (float*) alloc((size_t)BHc * Sc * 4);
  float*  amean = (float*) alloc(512);
  float*  emean = (float*) alloc(512);
  bf16_t* nW0T  = (bf16_t*)alloc((size_t)BHc * HDc * HIDc * 2);
  bf16_t* nW1T  = (bf16_t*)alloc((size_t)BHc * HDc * HIDc * 2);

  dim3 gemmGrid(Nrows / 128, Dc / 128);
  int convBlocks = (Dc * Dc + 255) / 256;

  // store path: x_n, keys, values
  k_rmsnorm_row<<<Nrows, 256, 0, stream>>>(x, snw, xsn);
  k_convT<<<convBlocks, 256, 0, stream>>>(W_K, wt, Dc, Dc);
  k_gemm_bf16<<<gemmGrid, 256, 0, stream>>>(xsn, wt, Cf, Nrows, Dc, Dc);
  k_split<<<dim3(Nrows, Hc), 64, 0, stream>>>(Cf, knw, keys);
  k_convT<<<convBlocks, 256, 0, stream>>>(W_V, wt, Dc, Dc);
  k_gemm_bf16<<<gemmGrid, 256, 0, stream>>>(xsn, wt, Cf, Nrows, Dc, Dc);
  k_split<<<dim3(Nrows, Hc), 64, 0, stream>>>(Cf, (const float*)nullptr, vals);

  // gates + means
  k_gates<<<Nrows, 64, 0, stream>>>(xsn, aw, ab, tw, tb, ew, eb, alp, tht, eta);
  k_mean<<<BHc, 256, 0, stream>>>(alp, eta, amean, emean);

  // retrieve path projection: q (xsn reused for retrieve-normed x)
  k_rmsnorm_row<<<Nrows, 256, 0, stream>>>(x, rnw, xsn);
  k_convT<<<convBlocks, 256, 0, stream>>>(W_Q, wt, Dc, Dc);
  k_gemm_bf16<<<gemmGrid, 256, 0, stream>>>(xsn, wt, Cf, Nrows, Dc, Dc);
  k_split<<<dim3(Nrows, Hc), 64, 0, stream>>>(Cf, qnw, qn);

  // fast-weight grad + update, then retrieval
  k_memgrad<<<BHc, 256, 0, stream>>>(keys, vals, tht, mW0, mW1, amean, nW0T, nW1T);
  k_retrieve<<<dim3(Sc / 32, BHc), 256, 0, stream>>>(qn, nW0T, nW1T, out);
}
